// CausalAttention_45801531244733
// MI455X (gfx1250) — matrix operand order repaired and checked by reading, not compiled, not run on hardware
//
#include <hip/hip_runtime.h>
#include <hip/hip_bf16.h>
#include <cstddef>

// ---------------------------------------------------------------------------
// CDNA5 (gfx1250) causal attention.
//   - all matmuls via v_wmma_f32_16x16x32_f16 (wave32 WMMA)
//   - GEMM A/B tiles and flash-attention K tiles DMA'd into LDS by the
//     Tensor Data Mover (tensor_load_to_lds + s_wait_tensorcnt), with the
//     GEMM double-buffered so the TDM overlaps the matrix pipe.
// Problem: B=1, S=4096, D=2048, H=16, hd=128 (all compile-time constants).
// ---------------------------------------------------------------------------

#define S_LEN 4096
#define DMODEL 2048
#define NHEADS 16
#define HD 128

typedef _Float16 h16;
typedef __attribute__((ext_vector_type(8)))  _Float16 v8h;
typedef __attribute__((ext_vector_type(16))) _Float16 v16h;
typedef __attribute__((ext_vector_type(8)))  float    v8f;
typedef __attribute__((ext_vector_type(4)))  unsigned int v4u;
typedef __attribute__((ext_vector_type(8)))  unsigned int v8u;

__device__ __forceinline__ v16h mk16(v8h lo, v8h hi) {
    v16h r;
#pragma unroll
    for (int i = 0; i < 8; ++i) { r[i] = lo[i]; r[i + 8] = hi[i]; }
    return r;
}

// ------------------- Tensor Data Mover: 2D tile -> LDS ----------------------
// D# per CDNA5 ISA. 2D tile of f16 (data_size=1 -> 2B):
//   tile_w elements per row, tile_h rows, row stride = stride elements.
// Wave-level op (EXEC ignored); caller gates to one wave and drains with
// s_wait_tensorcnt before the workgroup barrier that publishes the tile.
__device__ __forceinline__ void tdm_load_tile_f16(unsigned lds_byte_addr,
                                                  const void* gptr,
                                                  unsigned tile_w, unsigned tile_h,
                                                  unsigned stride_elems) {
    unsigned long long ga = (unsigned long long)(size_t)gptr;
    v4u g0;
    g0[0] = 1u;                                            // count=1 (valid D#)
    g0[1] = lds_byte_addr;                                 // lds_addr
    g0[2] = (unsigned)ga;                                  // global_addr[31:0]
    g0[3] = (unsigned)((ga >> 32) & 0x1FFFFFFu) | (2u << 30); // [56:32]|type=2
    v8u g1;
    g1[0] = 1u << 16;                                      // data_size=1 (2B)
    g1[1] = (tile_w & 0xFFFFu) << 16;                      // tensor_dim0[15:0]
    g1[2] = (tile_w >> 16) | ((tile_h & 0xFFFFu) << 16);   // dim0[31:16]|dim1[15:0]
    g1[3] = (tile_h >> 16) | (tile_w << 16);               // dim1[31:16]|tile_dim0
    g1[4] = tile_h & 0xFFFFu;                              // tile_dim1 (tile_dim2=0)
    g1[5] = stride_elems;                                  // tensor_dim0_stride[31:0]
    g1[6] = 0u;                                            // stride[47:32]|dim1_stride lo
    g1[7] = 0u;                                            // dim1_stride hi
    asm volatile("tensor_load_to_lds %0, %1" :: "s"(g0), "s"(g1) : "memory");
}

// --------------------------- f32 -> f16 convert -----------------------------
__global__ void cvt_f32_f16_kernel(const float* __restrict__ src,
                                   h16* __restrict__ dst, int n) {
    int i = (blockIdx.x * blockDim.x + threadIdx.x) * 4;
    if (i + 3 < n) {
        float4 v = *(const float4*)(src + i);
        dst[i + 0] = (h16)v.x; dst[i + 1] = (h16)v.y;
        dst[i + 2] = (h16)v.z; dst[i + 3] = (h16)v.w;
    }
}

// ----------------- transpose + convert W[K][N] -> Wt[N][K] f16 --------------
__global__ void transpose_cvt_kernel(const float* __restrict__ src,
                                     h16* __restrict__ dst, int dim) {
    __shared__ float tile[32][33];
    int bx = blockIdx.x * 32, by = blockIdx.y * 32;
    int tx = threadIdx.x, ty = threadIdx.y;   // 32 x 8
#pragma unroll
    for (int i = 0; i < 32; i += 8)
        tile[ty + i][tx] = src[(size_t)(by + ty + i) * dim + bx + tx];
    __syncthreads();
#pragma unroll
    for (int i = 0; i < 32; i += 8)
        dst[(size_t)(bx + ty + i) * dim + by + tx] = (h16)tile[tx][ty + i];
}

// --------------------------- RoPE (in place, f16) ---------------------------
__global__ void rope_kernel(h16* __restrict__ X) {
    int idx = blockIdx.x * blockDim.x + threadIdx.x;     // over S*D/2 pairs
    int s   = idx / (DMODEL / 2);
    int rem = idx % (DMODEL / 2);
    int hh  = rem >> 6;                                  // 64 pairs per head
    int p   = rem & 63;
    size_t e = (size_t)s * DMODEL + (size_t)hh * HD + 2 * p;
    float freq = __expf(-((float)(2 * p) / (float)HD) * 9.210340371976184f);
    float ang = (float)s * freq;
    float sn, cs;
    __sincosf(ang, &sn, &cs);
    float xr = (float)X[e], xi = (float)X[e + 1];
    X[e]     = (h16)(xr * cs - xi * sn);
    X[e + 1] = (h16)(xr * sn + xi * cs);
}

// ------------------- WMMA GEMM: C[M][N] = A[M][K] * Bt[N][K]^T --------------
// Fixed 4096x2048x2048. Block tile 128x128, 256 threads (8 waves), each wave
// 32x64 = 2x4 WMMA fragments; K-step 64 (two 32 sub-steps). Tiles arrive via
// double-buffered TDM DMA: wave 0 issues the next A/B tile loads, everyone
// computes on the current buffer, wave 0 drains TENSORcnt, barrier publishes.
constexpr int GM = S_LEN, GN = DMODEL, GK = DMODEL;

template <bool STORE_F32>
__global__ void __launch_bounds__(256)
gemm_wmma_kernel(const h16* __restrict__ A, const h16* __restrict__ Bt,
                 void* __restrict__ Cout) {
    __shared__ h16 sA[2][128 * 64];   // 2 x 16 KB
    __shared__ h16 sB[2][128 * 64];   // 2 x 16 KB
    const int tid  = threadIdx.x;
    const int lane = tid & 31, wid = tid >> 5;
    const int half = lane >> 4, ln = lane & 15;
    const int mBase = blockIdx.y * 128, nBase = blockIdx.x * 128;
    const int wm = wid & 3, wn = wid >> 2;      // 4 (M) x 2 (N) waves
    const bool tdmWave = (wid == 0);
    v8f acc[2][4] = {};

    // prologue: DMA first K-step tiles into buffer 0
    if (tdmWave) {
        tdm_load_tile_f16((unsigned)(size_t)(void*)&sA[0][0],
                          &A[(size_t)mBase * GK], 64, 128, GK);
        tdm_load_tile_f16((unsigned)(size_t)(void*)&sB[0][0],
                          &Bt[(size_t)nBase * GK], 64, 128, GK);
        __builtin_amdgcn_s_wait_tensorcnt(0);
    }
    __syncthreads();

    for (int kt = 0; kt < GK; kt += 64) {
        const int buf = (kt >> 6) & 1;
        // kick off DMA for the next K-step while we compute on this one
        if (tdmWave && (kt + 64 < GK)) {
            tdm_load_tile_f16((unsigned)(size_t)(void*)&sA[buf ^ 1][0],
                              &A[(size_t)mBase * GK + kt + 64], 64, 128, GK);
            tdm_load_tile_f16((unsigned)(size_t)(void*)&sB[buf ^ 1][0],
                              &Bt[(size_t)nBase * GK + kt + 64], 64, 128, GK);
        }

#pragma unroll
        for (int ks = 0; ks < 2; ++ks) {
            const int kc = ks * 32;
            v16h af[2], bf[4];
#pragma unroll
            for (int f = 0; f < 2; ++f) {
                // A 16x32 layout: lane row = ln, K segs at 8*half, 16+8*half
                int ar = wm * 32 + f * 16 + ln;
                af[f] = mk16(*(const v8h*)&sA[buf][ar * 64 + kc + 8 * half],
                             *(const v8h*)&sA[buf][ar * 64 + kc + 16 + 8 * half]);
            }
#pragma unroll
            for (int f = 0; f < 4; ++f) {
                // B 32x16 layout: lane col = ln, K segs at 8*half, 16+8*half
                int br = wn * 64 + f * 16 + ln;
                bf[f] = mk16(*(const v8h*)&sB[buf][br * 64 + kc + 8 * half],
                             *(const v8h*)&sB[buf][br * 64 + kc + 16 + 8 * half]);
            }
#pragma unroll
            for (int fm = 0; fm < 2; ++fm)
#pragma unroll
                for (int fn = 0; fn < 4; ++fn)
                    acc[fm][fn] = __builtin_amdgcn_wmma_f32_16x16x32_f16(
                        false, af[fm], false, bf[fn], (short)0, acc[fm][fn],
                        false, false);
        }

        // drain next-buffer DMA (wave-private TENSORcnt), then publish
        if (tdmWave) __builtin_amdgcn_s_wait_tensorcnt(0);
        __syncthreads();
    }

    // C/D layout: row = r + 8*half, col = ln; compile-time strides
#pragma unroll
    for (int fm = 0; fm < 2; ++fm)
#pragma unroll
        for (int fn = 0; fn < 4; ++fn)
#pragma unroll
            for (int r = 0; r < 8; ++r) {
                int row = mBase + wm * 32 + fm * 16 + r + 8 * half;
                int col = nBase + wn * 64 + fn * 16 + ln;
                if constexpr (STORE_F32)
                    ((float*)Cout)[(size_t)row * GN + col] = acc[fm][fn][r];
                else
                    ((h16*)Cout)[(size_t)row * GN + col] = (h16)acc[fm][fn][r];
            }
}

// ------------------------- Flash attention (causal) -------------------------
// Grid: (S/64, H). 4 waves per WG; wave w owns query rows q0+16w .. q0+16w+15.
// K tile (32 keys x 128 hd, row-major) arrives via the Tensor Data Mover;
// V tile is transposed into LDS by VALU copies. QK^T and PV via WMMA, online
// softmax in f32 with wave32 shfl reductions.
__global__ void __launch_bounds__(128)
flash_attn_kernel(const h16* __restrict__ Q, const h16* __restrict__ Kg,
                  const h16* __restrict__ Vg, h16* __restrict__ O) {
    __shared__ h16 sK[32 * HD];      // [key][hd]   (TDM destination)
    __shared__ h16 sV[HD * 32];      // [hd][key]   (transposed V tile)
    __shared__ h16 sP[4][16 * 32];   // per-wave P tile (C-layout -> A-layout)

    const int tid = threadIdx.x, lane = tid & 31, wid = tid >> 5;
    const int half = lane >> 4, ln = lane & 15;
    const int q0 = blockIdx.x * 64;
    const int h  = blockIdx.y;
    const size_t hoff = (size_t)h * HD;
    const unsigned sK_lds = (unsigned)(size_t)(void*)&sK[0];

    // Q fragments for this wave's 16 rows: 4 chunks of K=32 over hd=128
    v16h qf[4];
    {
        const h16* qrow = Q + (size_t)(q0 + wid * 16 + ln) * DMODEL + hoff;
#pragma unroll
        for (int kk = 0; kk < 4; ++kk)
            qf[kk] = mk16(*(const v8h*)&qrow[kk * 32 + 8 * half],
                          *(const v8h*)&qrow[kk * 32 + 16 + 8 * half]);
    }

    v8f o[8] = {};
    float mrow[8], lrow[8];
#pragma unroll
    for (int r = 0; r < 8; ++r) { mrow[r] = -1e30f; lrow[r] = 0.0f; }

    const int kEnd = q0 + 64;                      // causal bound for this WG
    for (int kb = 0; kb < kEnd; kb += 32) {
        // K tile via TDM (one wave issues; TENSORcnt is wave-private)
        if (wid == 0) {
            tdm_load_tile_f16(sK_lds, &Kg[(size_t)kb * DMODEL + hoff],
                              HD, 32, DMODEL);
            __builtin_amdgcn_s_wait_tensorcnt(0);
        }
        // V tile: cooperative load + transpose into [hd][key]
#pragma unroll
        for (int s4 = 0; s4 < 4; ++s4) {
            int seg = tid * 4 + s4;                // 0..511
            int row = seg >> 4;                    // key 0..31
            int col = (seg & 15) * 8;              // hd 0..120
            v8h w = *(const v8h*)&Vg[(size_t)(kb + row) * DMODEL + hoff + col];
#pragma unroll
            for (int i = 0; i < 8; ++i)
                sV[(col + i) * 32 + row] = w[i];
        }
        if (kb + 32 < kEnd)                        // global_prefetch_b8
            __builtin_prefetch(&Vg[(size_t)(kb + 32 + lane) * DMODEL + hoff], 0, 1);
        __syncthreads();

        // scores: 16 q-rows x 32 keys = two 16x16 C fragments, K-dim hd=128
        v8f sc[2] = {};
#pragma unroll
        for (int kk = 0; kk < 4; ++kk) {
#pragma unroll
            for (int g = 0; g < 2; ++g) {
                const h16* kr = &sK[(g * 16 + ln) * HD + kk * 32 + 8 * half];
                v16h bK = mk16(*(const v8h*)&kr[0], *(const v8h*)&kr[16]);
                sc[g] = __builtin_amdgcn_wmma_f32_16x16x32_f16(
                    false, qf[kk], false, bK, (short)0, sc[g], false, false);
            }
        }

        // scale + causal mask + online softmax (rows r+8*half, cols ln)
        const float scale = 0.08838834764831845f;  // 1/sqrt(128)
#pragma unroll
        for (int r = 0; r < 8; ++r) {
            int qAbs = q0 + wid * 16 + r + 8 * half;
#pragma unroll
            for (int g = 0; g < 2; ++g) {
                int kAbs = kb + g * 16 + ln;
                float v = sc[g][r] * scale;
                sc[g][r] = (kAbs > qAbs) ? -1e30f : v;
            }
            float t = fmaxf(sc[0][r], sc[1][r]);
#pragma unroll
            for (int off = 1; off < 16; off <<= 1)      // stays within half
                t = fmaxf(t, __shfl_xor(t, off, 32));
            float mnew = fmaxf(mrow[r], t);
            float corr = __expf(mrow[r] - mnew);
            float p0 = __expf(sc[0][r] - mnew);
            float p1 = __expf(sc[1][r] - mnew);
            float rs = p0 + p1;
#pragma unroll
            for (int off = 1; off < 16; off <<= 1)
                rs += __shfl_xor(rs, off, 32);
            lrow[r] = lrow[r] * corr + rs;
            mrow[r] = mnew;
#pragma unroll
            for (int n = 0; n < 8; ++n) o[n][r] *= corr;
            // stash P in C-layout; reload below in A-layout (same wave)
            sP[wid][(r + 8 * half) * 32 + ln]      = (h16)p0;
            sP[wid][(r + 8 * half) * 32 + 16 + ln] = (h16)p1;
        }

        // P as 16x32 A fragment (within-wave LDS roundtrip; compiler inserts
        // the s_wait_dscnt for its own ds ops)
        v16h pf = mk16(*(const v8h*)&sP[wid][ln * 32 + 8 * half],
                       *(const v8h*)&sP[wid][ln * 32 + 16 + 8 * half]);

        // O(16x128) += P(16x32) @ V(32x128): 8 column fragments
#pragma unroll
        for (int n = 0; n < 8; ++n) {
            const h16* vr = &sV[(n * 16 + ln) * 32 + 8 * half];
            v16h bV = mk16(*(const v8h*)&vr[0], *(const v8h*)&vr[16]);
            o[n] = __builtin_amdgcn_wmma_f32_16x16x32_f16(
                false, pf, false, bV, (short)0, o[n], false, false);
        }
        __syncthreads();
    }

    // normalize and store (attn output in [S][D] f16, feeds final GEMM)
#pragma unroll
    for (int r = 0; r < 8; ++r) {
        float inv = 1.0f / lrow[r];
        int row = q0 + wid * 16 + r + 8 * half;
#pragma unroll
        for (int n = 0; n < 8; ++n)
            O[(size_t)row * DMODEL + hoff + n * 16 + ln] =
                (h16)(o[n][r] * inv);
    }
}

// ---------------------------------------------------------------------------
extern "C" void kernel_launch(void* const* d_in, const int* in_sizes, int n_in,
                              void* d_out, int out_size, void* d_ws, size_t ws_size,
                              hipStream_t stream) {
    (void)in_sizes; (void)n_in; (void)out_size; (void)ws_size;
    const float* x  = (const float*)d_in[0];
    /* d_in[1] = mask: causal mask applied analytically in-kernel */
    const float* wq = (const float*)d_in[2];
    const float* wk = (const float*)d_in[3];
    const float* wv = (const float*)d_in[4];
    const float* wo = (const float*)d_in[5];
    float* out = (float*)d_out;

    const size_t SD = (size_t)S_LEN * DMODEL;   // 8M elems
    const size_t DD = (size_t)DMODEL * DMODEL;  // 4M elems

    // f16 workspace layout (5*SD + 4*DD halfs = 112 MB)
    h16* xb  = (h16*)d_ws;
    h16* wqT = xb  + SD;
    h16* wkT = wqT + DD;
    h16* wvT = wkT + DD;
    h16* woT = wvT + DD;
    h16* Qb  = woT + DD;
    h16* Kb  = Qb + SD;
    h16* Vb  = Kb + SD;
    h16* Ab  = Vb + SD;

    // 1) convert x to f16
    cvt_f32_f16_kernel<<<(int)(SD / 4 / 256), 256, 0, stream>>>(x, xb, (int)SD);

    // 2) transpose+convert weights to [N][K] f16
    dim3 tB(32, 8), tG(DMODEL / 32, DMODEL / 32);
    transpose_cvt_kernel<<<tG, tB, 0, stream>>>(wq, wqT, DMODEL);
    transpose_cvt_kernel<<<tG, tB, 0, stream>>>(wk, wkT, DMODEL);
    transpose_cvt_kernel<<<tG, tB, 0, stream>>>(wv, wvT, DMODEL);
    transpose_cvt_kernel<<<tG, tB, 0, stream>>>(wo, woT, DMODEL);

    // 3) Q/K/V projections (WMMA + double-buffered TDM)
    dim3 gG(GN / 128, GM / 128);
    gemm_wmma_kernel<false><<<gG, 256, 0, stream>>>(xb, wqT, Qb);
    gemm_wmma_kernel<false><<<gG, 256, 0, stream>>>(xb, wkT, Kb);
    gemm_wmma_kernel<false><<<gG, 256, 0, stream>>>(xb, wvT, Vb);

    // 4) RoPE on Q and K
    rope_kernel<<<(int)(SD / 2 / 256), 256, 0, stream>>>(Qb);
    rope_kernel<<<(int)(SD / 2 / 256), 256, 0, stream>>>(Kb);

    // 5) causal flash attention (WMMA + TDM)
    flash_attn_kernel<<<dim3(S_LEN / 64, NHEADS), 128, 0, stream>>>(Qb, Kb, Vb, Ab);

    // 6) output projection, f32 store (WMMA + double-buffered TDM)
    gemm_wmma_kernel<true><<<gG, 256, 0, stream>>>(Ab, woT, out);
}
